// Encoder_75935021793540
// MI455X (gfx1250) — matrix operand order repaired and checked by reading, not compiled, not run on hardware
//
#include <hip/hip_runtime.h>

// MI455X / gfx1250, wave32. Matrix work goes through v_wmma_f32_16x16x32_bf16
// (bf16 in, fp32 accumulate). Reference is fp32; bf16 GEMM inputs with fp32
// accumulation + fp32-resident hidden state is the hardware-rewarded path.

typedef __bf16 bf16_t;
typedef __attribute__((ext_vector_type(16))) __bf16 v16bf;
typedef __attribute__((ext_vector_type(8)))  float  v8f;

#define NB   32            // batch
#define NT   2048          // seq len
#define ND   256           // layer-1 input dim
#define NU   512           // hidden units
#define N3U  1536          // 3*U
#define MROWS (NB * NT)    // 65536 rows, row = b*NT + t

// ---------------------------------------------------------------------------
// Y[M][1536] = bf16(A[M][K]) x bf16(W[K][1536]) + bias   (fp32 out)
// One wave computes a 16x64 strip (4 n-tiles); 8 waves / block share the
// W tile, staged into a double-buffered LDS tile pre-swizzled into
// B-fragment order (one contiguous 32B LDS read per fragment). All four
// B-fragments are fetched into distinct registers before the four WMMAs so
// the XDL pipe sees back-to-back v_wmma issues instead of ds-wait stalls.
// ---------------------------------------------------------------------------
__global__ __launch_bounds__(256) void wmma_gemm_proj(
    const float* __restrict__ A, const float* __restrict__ W,
    const float* __restrict__ bias, float* __restrict__ Y, int K)
{
  __shared__ alignas(16) bf16_t ldsB[2][4 * 32 * 16];  // [buf][ntile][lane][e] 8KB
  const int tid  = threadIdx.x;
  const int lane = tid & 31;
  const int wav  = tid >> 5;                 // 0..7
  const int mt   = blockIdx.x * 8 + wav;     // 16-row tile index
  const int n0   = blockIdx.y * 64;          // 64-col strip
  const int nloc = lane & 15;
  const int kgrp = lane >> 4;                // 0/1
  const int mrow = mt * 16 + nloc;
  const int nk   = K >> 5;                   // 32-wide k-tiles

  auto stage = [&](int kk, int buf) {
    // stage W[kk..kk+31][n0..n0+63] -> bf16, B-fragment order
    for (int i = tid; i < 2048; i += 256) {
      int r = i >> 6;                        // k in tile (0..31)
      int c = i & 63;                        // col in strip
      float w = W[(size_t)(kk + r) * N3U + n0 + c];
      int ntl = c >> 4, n = c & 15, kg = r >> 4, e = r & 15;
      // B layout: lane = kg*16+n holds K = kg*16 + e  (e contiguous)
      ldsB[buf][((ntl * 32) + (kg * 16 + n)) * 16 + e] = (bf16_t)w;
    }
  };

  v8f acc[4] = {};

  stage(0, 0);
  __syncthreads();

  for (int kt = 0; kt < nk; ++kt) {
    const int cur = kt & 1;
    if (kt + 1 < nk) stage((kt + 1) << 5, cur ^ 1);  // overlap with math

    // A fragment, contiguous order: K = 16*kgrp + e
    const float* ap = A + (size_t)mrow * K + (kt << 5) + kgrp * 16;
    v16bf af;
#pragma unroll
    for (int e = 0; e < 8; ++e) af[e]     = (bf16_t)ap[e];
#pragma unroll
    for (int e = 0; e < 8; ++e) af[8 + e] = (bf16_t)ap[8 + e];

    v16bf bf4[4];
#pragma unroll
    for (int ntl = 0; ntl < 4; ++ntl)
      bf4[ntl] = *(const v16bf*)&ldsB[cur][(ntl * 32 + lane) * 16];
#pragma unroll
    for (int ntl = 0; ntl < 4; ++ntl)
      acc[ntl] = __builtin_amdgcn_wmma_f32_16x16x32_bf16(
          false, af, false, bf4[ntl], (short)0, acc[ntl], false, false);

    __syncthreads();  // iteration boundary: cur reads done, nxt writes done
  }

  // D layout: lane holds n = lane&15, rows m = vi + 8*(lane>>4)
#pragma unroll
  for (int ntl = 0; ntl < 4; ++ntl) {
    int col = n0 + ntl * 16 + nloc;
    float b = bias[col];
#pragma unroll
    for (int vi = 0; vi < 8; ++vi) {
      int row = mt * 16 + kgrp * 8 + vi;
      Y[(size_t)row * N3U + col] = acc[ntl][vi] + b;
    }
  }
}

// ---------------------------------------------------------------------------
// Persistent GRU scan. 16 blocks, each owns hidden columns [j0, j0+32) and
// their 3 gate columns. 12 waves/block = 2 M-tiles x 6 N-tiles; recurrent
// weights live in VGPRs as prebuilt B-fragments for the whole 2048-step scan.
// One grid barrier per step; hidden state double-buffered in global (bf16).
// Next step's xproj rows are prefetched (global_prefetch_b8) during the
// current step's matmul to shorten the sequential critical path.
// ---------------------------------------------------------------------------
__global__ __launch_bounds__(384) void gru_scan(
    const float* __restrict__ xproj,   // [MROWS][1536], row = b*NT + t
    const float* __restrict__ rk,      // [512][1536]
    const float* __restrict__ brec,    // [1536] recurrent bias
    const float* __restrict__ h0,      // [32][512] initial state
    float* __restrict__ out,           // [MROWS][512]
    float* __restrict__ state_out,     // [32][512]
    bf16_t* __restrict__ hbuf,         // 2 x [32][512] bf16 (double buffer)
    unsigned* __restrict__ bar)        // grid barrier counter (zeroed on host)
{
  __shared__ alignas(16) bf16_t h_lds[NB * NU];   // 32 KB staged state
  __shared__ float inner_s[NB * 96];              // 12 KB  h@rk result
  __shared__ float h_own[NB * 32];                // 4 KB   fp32-resident state

  const int tid  = threadIdx.x;
  const int lane = tid & 31;
  const int w    = tid >> 5;        // 0..11
  const int mt   = w & 1;           // M tile
  const int ntl  = w >> 1;          // 0..5 N tile
  const int gate = ntl >> 1;        // z / r / h
  const int half = ntl & 1;
  const int nloc = lane & 15;
  const int kgrp = lane >> 4;
  const int j0   = blockIdx.x * 32;

  // Pre-build 16 recurrent-weight B-fragments (K=512) in registers.
  v16bf bfrag[16];
  {
    const int col = gate * 512 + j0 + half * 16 + nloc;
    const float* rp = rk + col;
#pragma unroll
    for (int kt = 0; kt < 16; ++kt) {
#pragma unroll
      for (int e = 0; e < 16; ++e)
        bfrag[kt][e] = (bf16_t)rp[(size_t)(kt * 32 + kgrp * 16 + e) * N3U];
    }
  }

  // init own columns + double-buffer slot 0
  for (int i = tid; i < NB * 32; i += 384) {
    int m = i >> 5, u = i & 31;
    float h = h0[m * NU + j0 + u];
    h_own[i] = h;
    hbuf[m * NU + j0 + u] = (bf16_t)h;
  }

  for (int t = 0; t < NT; ++t) {
    // ---- grid barrier: release step t-1 writes, then all-arrive ----
    __threadfence();                       // every writer releases its stores
    __syncthreads();
    if (tid == 0) {
      __builtin_amdgcn_s_cluster_barrier();  // CDNA5 cluster sync (NOP if unclustered)
      atomicAdd(bar, 1u);
      const unsigned tgt = 16u * (unsigned)(t + 1);
      while (atomicAdd(bar, 0u) < tgt) { __builtin_amdgcn_s_sleep(1); }
    }
    __syncthreads();

    const bf16_t* hsrc = hbuf + (size_t)(t & 1) * (NB * NU);
    bf16_t*       hdst = hbuf + (size_t)((t + 1) & 1) * (NB * NU);

    // stage h (bf16) into LDS with agent-scope loads (no stale near-cache hits)
    for (int c = tid; c < (NB * NU) / 4; c += 384) {
      unsigned long long v = __hip_atomic_load(
          ((const unsigned long long*)hsrc) + c,
          __ATOMIC_RELAXED, __HIP_MEMORY_SCOPE_AGENT);
      ((unsigned long long*)h_lds)[c] = v;
    }

    // prefetch next timestep's gate inputs while the matmul runs
    if (t + 1 < NT) {
      for (int i = tid; i < NB * 32; i += 384) {
        int m = i >> 5, u = i & 31;
        const float* p = &xproj[((size_t)m * NT + (t + 1)) * N3U] + j0 + u;
        __builtin_prefetch(p, 0, 3);            // global_prefetch_b8
        __builtin_prefetch(p + 512, 0, 3);
        __builtin_prefetch(p + 1024, 0, 3);
      }
    }
    __syncthreads();

    // inner = h @ rk_slice  (one 16x16 tile per wave, K=512)
    v8f acc = {};
#pragma unroll
    for (int kt = 0; kt < 16; ++kt) {
      const bf16_t* apg = &h_lds[(mt * 16 + nloc) * NU + kt * 32 + kgrp * 16];
      union { uint4 q[2]; v16bf v; } au;
      au.q[0] = *(const uint4*)apg;        // K = 16*kgrp + 0..7
      au.q[1] = *(const uint4*)(apg + 8);  // K = 16*kgrp + 8..15
      acc = __builtin_amdgcn_wmma_f32_16x16x32_bf16(
          false, au.v, false, bfrag[kt], (short)0, acc, false, false);
    }

    // publish tile for the cross-gate elementwise update
#pragma unroll
    for (int vi = 0; vi < 8; ++vi)
      inner_s[(mt * 16 + kgrp * 8 + vi) * 96 + ntl * 16 + nloc] = acc[vi];
    __syncthreads();

    // gating + state update (fp32)
    for (int i = tid; i < NB * 32; i += 384) {
      int m = i >> 5, u = i & 31;
      int j = j0 + u;
      size_t xr0 = ((size_t)m * NT + t) * N3U;
      float xz = xproj[xr0 + j];
      float xr = xproj[xr0 + 512 + j];
      float xh = xproj[xr0 + 1024 + j];
      float iz = inner_s[m * 96 +      u] + brec[j];
      float ir = inner_s[m * 96 + 32 + u] + brec[512 + j];
      float ih = inner_s[m * 96 + 64 + u] + brec[1024 + j];
      float z  = 1.0f / (1.0f + __expf(-(xz + iz)));
      float r  = 1.0f / (1.0f + __expf(-(xr + ir)));
      float hh = tanhf(xh + r * ih);
      float hp = h_own[i];
      float hn = z * hp + (1.0f - z) * hh;
      h_own[i] = hn;
      hdst[m * NU + j] = (bf16_t)hn;
      out[((size_t)m * NT + t) * NU + j] = hn;
    }
  }

  __syncthreads();
  for (int i = tid; i < NB * 32; i += 384) {
    int m = i >> 5, u = i & 31;
    state_out[m * NU + j0 + u] = h_own[i];
  }
}

// ---------------------------------------------------------------------------
extern "C" void kernel_launch(void* const* d_in, const int* in_sizes, int n_in,
                              void* d_out, int out_size, void* d_ws, size_t ws_size,
                              hipStream_t stream)
{
  (void)in_sizes; (void)n_in; (void)out_size; (void)ws_size;
  const float* x      = (const float*)d_in[0];
  const float* hidden = (const float*)d_in[1];
  const float* k1     = (const float*)d_in[2];
  const float* rk1    = (const float*)d_in[3];
  const float* b1     = (const float*)d_in[4];   // [2][1536]
  const float* k2     = (const float*)d_in[5];
  const float* rk2    = (const float*)d_in[6];
  const float* b2     = (const float*)d_in[7];

  float* out2   = (float*)d_out;                       // [32][2048][512]
  float* state1 = out2 + (size_t)NB * NT * NU;
  float* state2 = state1 + (size_t)NB * NU;

  // workspace: xproj (384MB) | hbuf1 | hbuf2 | barrier counters
  char*   ws         = (char*)d_ws;
  float*  xproj      = (float*)ws;
  size_t  xprojBytes = (size_t)MROWS * N3U * sizeof(float);
  bf16_t* hbuf1      = (bf16_t*)(ws + xprojBytes);
  bf16_t* hbuf2      = hbuf1 + 2 * NB * NU;
  unsigned* bars     = (unsigned*)(ws + xprojBytes + 4 * NB * NU * sizeof(bf16_t));

  hipMemsetAsync(bars, 0, 256, stream);   // deterministic across graph replays

  dim3 gemmGrid(MROWS / 16 / 8, N3U / 64, 1);   // (512, 24)

  // Layer 1: xproj1 = x @ k1 + b1_in ; scan -> out1 (staged in out2 region)
  wmma_gemm_proj<<<gemmGrid, 256, 0, stream>>>(x, k1, b1, xproj, ND);
  gru_scan<<<16, 384, 0, stream>>>(xproj, rk1, b1 + N3U, hidden,
                                   out2, state1, hbuf1, bars + 0);

  // Layer 2: xproj2 = out1 @ k2 + b2_in ; scan overwrites out2 in place
  wmma_gemm_proj<<<gemmGrid, 256, 0, stream>>>(out2, k2, b2, xproj, NU);
  gru_scan<<<16, 384, 0, stream>>>(xproj, rk2, b2 + N3U, hidden,
                                   out2, state2, hbuf2, bars + 16);
}
